// CausalSelfAttention_64372969832534
// MI455X (gfx1250) — matrix operand order repaired and checked by reading, not compiled, not run on hardware
//
#include <hip/hip_runtime.h>
#include <hip/hip_bf16.h>

typedef __attribute__((ext_vector_type(16))) __bf16 v16bf;
typedef __attribute__((ext_vector_type(8)))  float  v8f;

#define B_  2
#define N_  2048
#define D_  1024
#define H_  16
#define DH_ 64
#define E3_ 192   // 3*DH (k,q,v chunks)

// ---- WMMA fragment helpers (this code's own k order: k = 16*half + i in both A and B) ----

// A-matrix 16x32 bf16: lane idx = row M; half=0 -> K 0..15, half=1 -> K 16..31
__device__ __forceinline__ v16bf load_fragA(const __bf16* rowbase, int half) {
  v16bf f;
  const int kb = half * 16;
#pragma unroll
  for (int i = 0; i < 16; ++i) f[i] = rowbase[kb + i];
  return f;
}

// B-matrix 32x16 bf16: lane idx = col N; lanes 0-15 hold K=0..15, lanes 16-31 hold K=16..31
__device__ __forceinline__ v16bf load_fragB(const __bf16* rowbase, int half) {
  v16bf f;
  const int kb = half * 16;
#pragma unroll
  for (int i = 0; i < 16; ++i) f[i] = rowbase[kb + i];
  return f;
}

__device__ __forceinline__ v8f wmma_bf16(v16bf a, v16bf b, v8f c) {
  return __builtin_amdgcn_wmma_f32_16x16x32_bf16(false, a, false, b, (short)0, c,
                                                 false, false);
}

// ---- f32 -> bf16 conversion (grid-stride) ----
__global__ void cvt_kernel(const float* __restrict__ src, __bf16* __restrict__ dst, int n) {
  for (int i = blockIdx.x * blockDim.x + threadIdx.x; i < n; i += gridDim.x * blockDim.x)
    dst[i] = (__bf16)src[i];
}

// ---- QKV projection: one wave = 32 rows x one 64-wide chunk (k, q, or v) ----
// Register-blocked 2x4 tiles: per k-step 2 A-frags + 4 B-frags feed 8 WMMAs.
__global__ void __launch_bounds__(128) qkv_kernel(const __bf16* __restrict__ xb,
                                                  const __bf16* __restrict__ wqb,
                                                  const float*  __restrict__ bqkv,
                                                  __bf16* __restrict__ Kb,
                                                  __bf16* __restrict__ Qb,
                                                  __bf16* __restrict__ Vt) {
  const int lane = threadIdx.x & 31;
  const int wave = threadIdx.x >> 5;
  const int half = lane >> 4;
  const int idx  = lane & 15;
  const int wid  = blockIdx.x * 4 + wave;           // 6144 waves total
  const int rg    = wid & 63;                       // 64 row groups of 32
  const int chunk = (wid >> 6) % 3;                 // 0=k, 1=q, 2=v (wave-uniform)
  const int bh    = wid / 192;
  const int b = bh >> 4, h = bh & (H_ - 1);
  const int n0 = rg * 32;
  const int e0 = chunk * 64;

  const __bf16* arow0 = xb + ((size_t)b * N_ + n0 + idx) * D_;
  const __bf16* arow1 = arow0 + (size_t)16 * D_;
  const __bf16* brow[4];
#pragma unroll
  for (int t = 0; t < 4; ++t)
    brow[t] = wqb + ((size_t)h * E3_ + e0 + t * 16 + idx) * D_;

  v8f acc[2][4] = {{{}, {}, {}, {}}, {{}, {}, {}, {}}};
#pragma unroll 2
  for (int k0 = 0; k0 < D_; k0 += 32) {
    const v16bf a0 = load_fragA(arow0 + k0, half);
    const v16bf a1 = load_fragA(arow1 + k0, half);
#pragma unroll
    for (int t = 0; t < 4; ++t) {
      const v16bf bf = load_fragB(brow[t] + k0, half);
      acc[0][t] = wmma_bf16(a0, bf, acc[0][t]);
      acc[1][t] = wmma_bf16(a1, bf, acc[1][t]);
    }
  }

  float bias[4];
#pragma unroll
  for (int t = 0; t < 4; ++t) bias[t] = bqkv[h * E3_ + e0 + t * 16 + idx];

#pragma unroll
  for (int rt = 0; rt < 2; ++rt)
#pragma unroll
    for (int t = 0; t < 4; ++t)
#pragma unroll
      for (int r = 0; r < 8; ++r) {
        const int   n  = n0 + rt * 16 + r + half * 8;
        const int   dh = t * 16 + idx;
        const float v  = acc[rt][t][r] + bias[t];
        if (chunk == 0)      Kb[((size_t)bh * N_ + n) * DH_ + dh] = (__bf16)v;
        else if (chunk == 1) Qb[((size_t)bh * N_ + n) * DH_ + dh] = (__bf16)v;
        else                 Vt[((size_t)bh * DH_ + dh) * N_ + n] = (__bf16)v;
      }
}

// ---- Flash attention: one wave per (b,h,16-query tile); online softmax; 32 keys/step ----
__global__ void __launch_bounds__(128) attn_kernel(const __bf16* __restrict__ Qb,
                                                   const __bf16* __restrict__ Kb,
                                                   const __bf16* __restrict__ Vt,
                                                   __bf16* __restrict__ sab) {
  __shared__ float  sS[4][16][33];   // raw scores, padded
  __shared__ __bf16 sP[4][16][32];   // probabilities (A-matrix staging)
  __shared__ float  sF[4][16];       // per-row rescale factors

  const int lane = threadIdx.x & 31;
  const int w    = threadIdx.x >> 5;
  const int half = lane >> 4;
  const int idx  = lane & 15;
  const int wid  = blockIdx.x * 4 + w;
  const int rt = wid & (N_ / 16 - 1);           // query row tile
  const int bh = wid >> 7;                      // N_/16 == 128
  const int b  = bh >> 4, h = bh & 15;
  const int q0 = rt * 16;

  const __bf16* qrow = Qb + ((size_t)bh * N_ + q0 + idx) * DH_;
  const v16bf aq0 = load_fragA(qrow, half);
  const v16bf aq1 = load_fragA(qrow + 32, half);

  v8f   acc[4] = {{}, {}, {}, {}};
  float m_i = -1e30f, l_i = 0.f;
  const int gq   = q0 + idx;                    // global query row (lanes 0-15)
  const int nblk = (q0 + 16 + 31) >> 5;         // 32-key blocks needed (causal)

  for (int jb = 0; jb < nblk; ++jb) {
    const int j0 = jb * 32;
    // S = Q K^T for 32 keys (two 16x16 tiles, 2 k-steps each over DH=64)
#pragma unroll
    for (int t = 0; t < 2; ++t) {
      const __bf16* krow = Kb + ((size_t)bh * N_ + j0 + t * 16 + idx) * DH_;
      v8f s = {};
      s = wmma_bf16(aq0, load_fragB(krow, half), s);
      s = wmma_bf16(aq1, load_fragB(krow + 32, half), s);
#pragma unroll
      for (int r = 0; r < 8; ++r)
        sS[w][r + half * 8][t * 16 + idx] = s[r] * 0.125f;   // 1/sqrt(DH)
    }
    asm volatile("s_wait_dscnt 0" ::: "memory");

    if (lane < 16) {                            // lanes 0-15: per-row online softmax
      float sv[32], mx = -1e30f;
#pragma unroll
      for (int c2 = 0; c2 < 32; ++c2) {
        const float s = (j0 + c2 <= gq) ? sS[w][lane][c2] : -1e30f;
        sv[c2] = s;
        mx = fmaxf(mx, s);
      }
      const float m_new = fmaxf(m_i, mx);
      const float alpha = __expf(m_i - m_new);
      float lsum = l_i * alpha;
#pragma unroll
      for (int c2 = 0; c2 < 32; ++c2) {
        const float p = __expf(sv[c2] - m_new);
        lsum += p;
        sP[w][lane][c2] = (__bf16)p;
      }
      m_i = m_new;
      l_i = lsum;
      sF[w][lane] = alpha;
    }
    asm volatile("s_wait_dscnt 0" ::: "memory");

    float al[8];
#pragma unroll
    for (int r = 0; r < 8; ++r) al[r] = sF[w][r + half * 8];
#pragma unroll
    for (int r = 0; r < 8; ++r) {
      acc[0][r] *= al[r]; acc[1][r] *= al[r]; acc[2][r] *= al[r]; acc[3][r] *= al[r];
    }
    // acc += P @ V  (A: 16x32 probs from LDS; B: Vt rows are contiguous over keys)
    const v16bf ap = load_fragA(&sP[w][idx][0], half);
#pragma unroll
    for (int t = 0; t < 4; ++t) {
      const __bf16* vrow = Vt + ((size_t)bh * DH_ + t * 16 + idx) * N_ + j0;
      acc[t] = wmma_bf16(ap, load_fragB(vrow, half), acc[t]);
    }
  }

  if (lane < 16) sF[w][lane] = 1.f / l_i;
  asm volatile("s_wait_dscnt 0" ::: "memory");
  float inv[8];
#pragma unroll
  for (int r = 0; r < 8; ++r) inv[r] = sF[w][r + half * 8];
#pragma unroll
  for (int t = 0; t < 4; ++t)
#pragma unroll
    for (int r = 0; r < 8; ++r) {
      const int n = q0 + r + half * 8;
      sab[((size_t)b * N_ + n) * D_ + h * DH_ + t * 16 + idx] = (__bf16)(acc[t][r] * inv[r]);
    }
}

// ---- Output projection: one wave = 32 rows x 64 cols (2x4 register blocking) ----
__global__ void __launch_bounds__(128) proj_kernel(const __bf16* __restrict__ sab,
                                                   const __bf16* __restrict__ wpb,
                                                   const float*  __restrict__ bp,
                                                   float* __restrict__ out) {
  const int lane = threadIdx.x & 31;
  const int wave = threadIdx.x >> 5;
  const int half = lane >> 4;
  const int idx  = lane & 15;
  const int wid  = blockIdx.x * 4 + wave;       // 2048 waves total
  const int JG = D_ / 64;                       // 16 col groups
  const int mg = wid / JG, jg = wid % JG;
  const int m0 = mg * 32, j0 = jg * 64;

  const __bf16* arow0 = sab + ((size_t)(m0 + idx)) * D_;
  const __bf16* arow1 = arow0 + (size_t)16 * D_;
  const __bf16* brow[4];
#pragma unroll
  for (int t = 0; t < 4; ++t)
    brow[t] = wpb + ((size_t)(j0 + t * 16 + idx)) * D_;

  v8f acc[2][4] = {{{}, {}, {}, {}}, {{}, {}, {}, {}}};
#pragma unroll 2
  for (int k0 = 0; k0 < D_; k0 += 32) {
    const v16bf a0 = load_fragA(arow0 + k0, half);
    const v16bf a1 = load_fragA(arow1 + k0, half);
#pragma unroll
    for (int t = 0; t < 4; ++t) {
      const v16bf bf = load_fragB(brow[t] + k0, half);
      acc[0][t] = wmma_bf16(a0, bf, acc[0][t]);
      acc[1][t] = wmma_bf16(a1, bf, acc[1][t]);
    }
  }

  float bias[4];
#pragma unroll
  for (int t = 0; t < 4; ++t) bias[t] = bp[j0 + t * 16 + idx];

#pragma unroll
  for (int rt = 0; rt < 2; ++rt)
#pragma unroll
    for (int t = 0; t < 4; ++t)
#pragma unroll
      for (int r = 0; r < 8; ++r)
        out[((size_t)(m0 + rt * 16 + r + half * 8)) * D_ + j0 + t * 16 + idx] =
            acc[rt][t][r] + bias[t];
}

extern "C" void kernel_launch(void* const* d_in, const int* in_sizes, int n_in,
                              void* d_out, int out_size, void* d_ws, size_t ws_size,
                              hipStream_t stream) {
  (void)in_sizes; (void)n_in; (void)out_size; (void)ws_size;
  const float* x     = (const float*)d_in[0];
  const float* Wqkv  = (const float*)d_in[1];
  const float* bqkv  = (const float*)d_in[2];
  const float* Wproj = (const float*)d_in[3];
  const float* bproj = (const float*)d_in[4];

  char* ws = (char*)d_ws;
  __bf16* xb  = (__bf16*)(ws);                      //  8 MB  x bf16
  __bf16* wqb = (__bf16*)(ws + ((size_t)8  << 20)); //  6 MB  Wqkv bf16
  __bf16* wpb = (__bf16*)(ws + ((size_t)14 << 20)); //  2 MB  Wproj bf16
  __bf16* Kb  = (__bf16*)(ws + ((size_t)16 << 20)); //  8 MB  K [BH,N,DH]
  __bf16* Qb  = (__bf16*)(ws + ((size_t)24 << 20)); //  8 MB  Q [BH,N,DH]
  __bf16* Vt  = (__bf16*)(ws + ((size_t)32 << 20)); //  8 MB  V^T [BH,DH,N]
  __bf16* sab = (__bf16*)(ws + ((size_t)40 << 20)); //  8 MB  attention out [B*N,D]

  cvt_kernel<<<2048, 256, 0, stream>>>(x, xb, B_ * N_ * D_);
  cvt_kernel<<<2048, 256, 0, stream>>>(Wqkv, wqb, H_ * E3_ * D_);
  cvt_kernel<<<1024, 256, 0, stream>>>(Wproj, wpb, D_ * D_);

  // 2*16 bh * 3 chunks * 64 row-groups = 6144 waves
  qkv_kernel<<<(B_ * H_ * 3 * (N_ / 32)) / 4, 128, 0, stream>>>(
      xb, wqb, bqkv, Kb, Qb, Vt);

  attn_kernel<<<(B_ * H_ * (N_ / 16)) / 4, 128, 0, stream>>>(Qb, Kb, Vt, sab);

  // (4096/32) row groups * (1024/64) col groups = 2048 waves
  proj_kernel<<<((B_ * N_ / 32) * (D_ / 64)) / 4, 128, 0, stream>>>(
      sab, wpb, bproj, (float*)d_out);
}
